// ConvLSTMCell_11751030522473
// MI455X (gfx1250) — compile-verified, matrix operand order checked by reading, not run on hardware
//
#include <hip/hip_runtime.h>
#include <hip/hip_bf16.h>

typedef __bf16 bf16;
typedef __attribute__((ext_vector_type(16))) __bf16 v16bf;
typedef __attribute__((ext_vector_type(8)))  float  v8f;

#define HID    64
#define OC     256
#define IMG    64
#define BATCH  4
#define TSTEPS 16
#define TAPBYTES (OC * HID * 2)   // 32768 bytes per tap of bf16 weights

// LDS byte offset of a shared-memory pointer (generic -> AS(3) -> u32)
__device__ __forceinline__ unsigned lds_off(const void* p) {
    return (unsigned)(unsigned long long)
           (__attribute__((address_space(3))) const void*)p;
}

// One per-lane async 16B copy: LDS[lds_byte] = gbase[goff .. goff+15]
// GLOBAL_LOAD_ASYNC_TO_LDS_B128, GVS addressing (saddr sgpr64 + vaddr u32).
// Tracked by ASYNCcnt, data bypasses VGPRs.
__device__ __forceinline__ void async_ld_lds_b128(unsigned lds_byte,
                                                  const void* gbase,
                                                  unsigned goff) {
    asm volatile("global_load_async_to_lds_b128 %0, %1, %2"
                 :
                 : "v"(lds_byte), "v"(goff), "s"(gbase)
                 : "memory");
}

// ---------------------------------------------------------------------------
// Pack fp32 weights [256][64][3][3] (OIHW) -> bf16 [tap][oc][ic], tap = ky*3+kx
// ---------------------------------------------------------------------------
__global__ __launch_bounds__(256)
void pack_weights(const float* __restrict__ w, bf16* __restrict__ wbf) {
    int i = blockIdx.x * 256 + threadIdx.x;
    if (i >= 9 * OC * HID) return;
    int tap = i / (OC * HID);
    int rem = i % (OC * HID);
    int oc = rem / HID, ic = rem % HID;
    int ky = tap / 3, kx = tap % 3;
    wbf[i] = (bf16)w[((oc * HID + ic) * 3 + ky) * 3 + kx];
}

// ---------------------------------------------------------------------------
// Zero initial hidden state (bf16 NHWC) and cell state (fp32 NCHW)
// ---------------------------------------------------------------------------
__global__ __launch_bounds__(256)
void zero_state(bf16* __restrict__ h0, float* __restrict__ c) {
    int i = blockIdx.x * 256 + threadIdx.x;
    if (i < BATCH * HID * IMG * IMG) {
        h0[i] = (bf16)0.f;
        c[i]  = 0.f;
    }
}

// ---------------------------------------------------------------------------
// One ConvLSTM time step.
//  - implicit GEMM: out[oc, px] = sum_{tap, ic} W[tap][oc][ic] * Hhalo[tap, ic, px]
//  - block tile: 256 oc x 32 px; 8 waves, wave = 32 oc x 32 px (2x2 WMMA tiles)
//  - weights streamed per-tap with global_load_async_to_lds_b128 (ASYNCcnt),
//    double buffered, overlapped with v_wmma_f32_16x16x32_bf16 compute
//  - gate-staging LDS aliases the weight buffers (dead after the tap loop)
//    to keep the block at ~77KB LDS -> 4 blocks/WGP
// ---------------------------------------------------------------------------
__global__ __launch_bounds__(256)
void convlstm_step(const float* __restrict__ xt,    // [B][256][64][64] slice at t
                   const bf16*  __restrict__ wbf,   // [9][256][64]
                   const bf16*  __restrict__ h_in,  // NHWC [B][64][64][64ic]
                   bf16*        __restrict__ h_out, // NHWC
                   float*       __restrict__ c_buf, // NCHW [B][64][64][64]
                   float*       __restrict__ out_t) // [B][64][64][64] slice at t
{
    // 64KB double-buffered tap weights + 13KB halo; sG (32KB) aliases sW,
    // which is dead once the tap loop's final barrier has completed.
    __shared__ __align__(16) char smem[2 * TAPBYTES + 3 * 34 * HID * 2];
    bf16*  const sW = (bf16*)smem;                       // [2][OC*HID]
    bf16*  const sH = (bf16*)(smem + 2 * TAPBYTES);      // [3][34][HID]
    float* const sG = (float*)smem;                      // [OC][32], aliases sW

    const int tid   = threadIdx.x;
    const int lane  = tid & 31;
    const int wave  = tid >> 5;
    const int pxt   = blockIdx.x;            // 0..127
    const int b     = blockIdx.y;            // 0..3
    const int y     = pxt >> 1;              // image row
    const int xbase = (pxt & 1) * 32;        // pixel tile start

    const unsigned sW_base = lds_off(sW);

    // ---- kick off async staging of tap 0 weights into buffer 0 ----
    {
        const unsigned l = sW_base + (unsigned)tid * 16u;
        const unsigned g = (unsigned)tid * 16u;
        #pragma unroll
        for (int i = 0; i < 8; ++i)
            async_ld_lds_b128(l + i * 4096u, wbf, g + i * 4096u);
    }

    // ---- prefetch the x_t slab this block consumes in the epilogue ----
    __builtin_prefetch(xt + ((b * OC + tid) * IMG + y) * IMG + xbase, 0, 1);

    // ---- stage 3x34 halo of h (ic-contiguous) into LDS ----
    for (int s = tid; s < 102 * 8; s += 256) {
        int chunk = s >> 3, seg = s & 7;     // 128B per (row,col), 8x16B segs
        int r  = chunk / 34, cc = chunk % 34;
        int gy = y + r - 1;
        int gx = xbase + cc - 1;
        uint4 v = make_uint4(0u, 0u, 0u, 0u);
        if (gy >= 0 && gy < IMG && gx >= 0 && gx < IMG) {
            v = *((const uint4*)(h_in + ((b * IMG + gy) * IMG + gx) * HID) + seg);
        }
        *((uint4*)(sH + (r * 34 + cc) * HID) + seg) = v;
    }

    // ---- accumulators: wave covers oc [wave*32, wave*32+32) x px [0,32) ----
    v8f acc[2][2] = {};

    const int ocw   = wave * 32;
    const int mrow  = lane & 15;             // M (or N) index within 16-tile
    const int khalf = (lane >> 4) * 16;      // K half selected by lane half

    #pragma unroll
    for (int tap = 0; tap < 9; ++tap) {
        const bf16* sWc = sW + (tap & 1) * (OC * HID);

        if (tap < 8) {
            // async-stage next tap's 32KB into the other buffer (overlaps WMMA)
            const unsigned l = sW_base + (unsigned)((tap + 1) & 1) * (unsigned)TAPBYTES
                             + (unsigned)tid * 16u;
            const unsigned g = (unsigned)(tap + 1) * (unsigned)TAPBYTES
                             + (unsigned)tid * 16u;
            #pragma unroll
            for (int i = 0; i < 8; ++i)
                async_ld_lds_b128(l + i * 4096u, wbf, g + i * 4096u);
            // 16 outstanding -> wait to 8: current tap's batch has landed
            asm volatile("s_wait_asynccnt 0x8" ::: "memory");
        } else {
            asm volatile("s_wait_asynccnt 0x0" ::: "memory");
        }
        __syncthreads();                     // all waves' slices visible

        const int ky = tap / 3, kx = tap % 3;
        #pragma unroll
        for (int chunk = 0; chunk < 2; ++chunk) {
            const int kk = chunk * 32 + khalf;       // ic base for this lane
            // A fragments: 16x32 bf16, A[m=oc][k=ic]; 32B contiguous per lane
            v16bf a0 = *(const v16bf*)(sWc + (ocw +      mrow) * HID + kk);
            v16bf a1 = *(const v16bf*)(sWc + (ocw + 16 + mrow) * HID + kk);
            // B fragments: 32x16 bf16, B[k=ic][n=px]; halo col = n + kx
            v16bf b0 = *(const v16bf*)(sH + (ky * 34 + (mrow      + kx)) * HID + kk);
            v16bf b1 = *(const v16bf*)(sH + (ky * 34 + (mrow + 16 + kx)) * HID + kk);
            acc[0][0] = __builtin_amdgcn_wmma_f32_16x16x32_bf16(false, a0, false, b0, (short)0, acc[0][0], false, false);
            acc[0][1] = __builtin_amdgcn_wmma_f32_16x16x32_bf16(false, a0, false, b1, (short)0, acc[0][1], false, false);
            acc[1][0] = __builtin_amdgcn_wmma_f32_16x16x32_bf16(false, a1, false, b0, (short)0, acc[1][0], false, false);
            acc[1][1] = __builtin_amdgcn_wmma_f32_16x16x32_bf16(false, a1, false, b1, (short)0, acc[1][1], false, false);
        }
        __syncthreads();   // everyone done reading before next async overwrite
    }

    // ---- epilogue: tmp = conv + x_t, stage all 4 gates to LDS ----
    // (sG aliases sW; the barrier above guarantees all sW reads are done)
    const int hi8 = (lane >> 4) * 8;         // C layout: lane half selects M+8
    #pragma unroll
    for (int mt = 0; mt < 2; ++mt) {
        #pragma unroll
        for (int nt = 0; nt < 2; ++nt) {
            const int px = nt * 16 + (lane & 15);
            #pragma unroll
            for (int j = 0; j < 8; ++j) {
                const int oc = ocw + mt * 16 + hi8 + j;
                float v = acc[mt][nt][j] +
                          xt[((b * OC + oc) * IMG + y) * IMG + xbase + px];
                sG[oc * 32 + px] = v;
            }
        }
    }
    __syncthreads();

    // ---- pointwise LSTM update: 64 ch x 32 px ----
    #pragma unroll
    for (int k = 0; k < 8; ++k) {
        int idx = k * 256 + tid;
        int ch = idx >> 5, px = idx & 31;
        float ci = sG[(0 * HID + ch) * 32 + px];
        float cf = sG[(1 * HID + ch) * 32 + px];
        float co = sG[(2 * HID + ch) * 32 + px];
        float cg = sG[(3 * HID + ch) * 32 + px];
        float ig = 1.f / (1.f + __expf(-ci));
        float fg = 1.f / (1.f + __expf(-cf));
        float og = 1.f / (1.f + __expf(-co));
        float gg = tanhf(cg);
        int cidx = ((b * HID + ch) * IMG + y) * IMG + xbase + px;
        float cn = fg * c_buf[cidx] + ig * gg;
        float hn = og * tanhf(cn);
        c_buf[cidx] = cn;
        out_t[cidx] = hn;
        h_out[((b * IMG + y) * IMG + xbase + px) * HID + ch] = (bf16)hn;
    }
}

// ---------------------------------------------------------------------------
// Host launcher
//   d_in[0] = x     [16,4,256,64,64] fp32
//   d_in[1] = w_h2h [256,64,3,3]     fp32
//   d_out   = hs    [16,4,64,64,64]  fp32
// Workspace layout (bytes):
//   [0,       288K) : bf16 packed weights
//   [1M,      3M)   : h ping (bf16 NHWC, 2MB)
//   [3M,      5M)   : h pong
//   [5M,      9M)   : c (fp32, 4MB)
// ---------------------------------------------------------------------------
extern "C" void kernel_launch(void* const* d_in, const int* in_sizes, int n_in,
                              void* d_out, int out_size, void* d_ws, size_t ws_size,
                              hipStream_t stream) {
    const float* x = (const float*)d_in[0];
    const float* w = (const float*)d_in[1];
    float* out = (float*)d_out;
    char*  ws  = (char*)d_ws;

    bf16*  wbf = (bf16*)(ws);
    bf16*  h0  = (bf16*)(ws + (size_t)(1 << 20));
    bf16*  h1  = (bf16*)(ws + (size_t)(3 << 20));
    float* c   = (float*)(ws + (size_t)(5 << 20));

    pack_weights<<<(9 * OC * HID + 255) / 256, 256, 0, stream>>>(w, wbf);
    zero_state<<<(BATCH * HID * IMG * IMG + 255) / 256, 256, 0, stream>>>(h0, c);

    bf16* hin = h0;
    bf16* hout = h1;
    const size_t x_step   = (size_t)BATCH * OC  * IMG * IMG;
    const size_t out_step = (size_t)BATCH * HID * IMG * IMG;
    for (int t = 0; t < TSTEPS; ++t) {
        dim3 grid(128, BATCH);
        convlstm_step<<<grid, 256, 0, stream>>>(x + (size_t)t * x_step, wbf,
                                                hin, hout, c,
                                                out + (size_t)t * out_step);
        bf16* tmp = hin; hin = hout; hout = tmp;
    }
}
